// LinearMixed_52570399703264
// MI455X (gfx1250) — matrix operand order repaired and checked by reading, not compiled, not run on hardware
//
#include <hip/hip_runtime.h>
#include <hip/hip_bf16.h>
#include <hip/hip_fp16.h>

#define OUT_FEATURES 11008
#define IN_FEATURES  4096
#define GROUP_SIZE   512

#define BM 128
#define BN 256
#define BK 64
#define LDS_STRIDE 72   // BK + 8 bf16 elements -> 144B row stride, conflict-free b128 reads
#define NSTAGES (IN_FEATURES / BK)

typedef __attribute__((ext_vector_type(16))) __bf16 v16bf;
typedef __attribute__((ext_vector_type(8)))  __bf16 v8bf;
typedef __attribute__((ext_vector_type(8)))  float  v8f;

// pack two fp32 -> one dword of two bf16 (round-to-nearest-even), a = low half
__device__ __forceinline__ unsigned pack2_bf16(float a, float b) {
    unsigned ua = __float_as_uint(a);
    unsigned ub = __float_as_uint(b);
    ua += 0x7fffu + ((ua >> 16) & 1u);
    ub += 0x7fffu + ((ub >> 16) & 1u);
    return (ua >> 16) | (ub & 0xffff0000u);
}

// A fragment (16x32 bf16): lane<16 -> K {kb..kb+7, kb+16..kb+23}; lane>=16 -> +8
__device__ __forceinline__ v16bf load_frag_a(const __bf16* row, int kb, int khalf) {
    v8bf lo = *(const v8bf*)(row + kb + khalf * 8);
    v8bf hi = *(const v8bf*)(row + kb + 16 + khalf * 8);
    return __builtin_shufflevector(lo, hi, 0,1,2,3,4,5,6,7,8,9,10,11,12,13,14,15);
}

// B fragment (32x16 bf16): lane<16 -> K {kb..kb+7, kb+16..kb+23}; lane>=16 -> +8
__device__ __forceinline__ v16bf load_frag_b(const __bf16* row, int kb, int khalf) {
    const __bf16* p = row + kb + khalf * 8;
    v8bf lo = *(const v8bf*)(p);
    v8bf hi = *(const v8bf*)(p + 16);
    return __builtin_shufflevector(lo, hi, 0,1,2,3,4,5,6,7,8,9,10,11,12,13,14,15);
}

// Stage one K-chunk: x fp32 -> bf16 into Asb (128x64), 4-bit W -> bf16 into Bsb (256x64).
__device__ __forceinline__ void stage_tiles(__bf16* __restrict__ Asb,
                                            __bf16* __restrict__ Bsb,
                                            const float* __restrict__ xRow,
                                            const unsigned char* __restrict__ wRow,
                                            const _Float16* __restrict__ wrange,
                                            const _Float16* __restrict__ wmin,
                                            int oRow, int ldRow, int ldKs, int tid, int kk) {
    // ---- A: 2 threads per row, 32 k-values each; fp32 -> bf16 ----
    {
        __bf16* dst = Asb + ldRow * LDS_STRIDE + ldKs;
#pragma unroll
        for (int i = 0; i < 4; ++i) {
            float4 v = *(const float4*)(xRow + kk + ldKs + i * 8);
            float4 w = *(const float4*)(xRow + kk + ldKs + i * 8 + 4);
            uint4 p;
            p.x = pack2_bf16(v.x, v.y);
            p.y = pack2_bf16(v.z, v.w);
            p.z = pack2_bf16(w.x, w.y);
            p.w = pack2_bf16(w.z, w.w);
            *(uint4*)(dst + i * 8) = p;
        }
    }
    // ---- B: 1 thread per out-feature row, all 64 k-values (one quant group) ----
    {
        const int g  = oRow * (IN_FEATURES / GROUP_SIZE) + (kk >> 9);
        const float s  = (float)wrange[g] * (1.0f / 15.0f);
        const float mn = (float)wmin[g];
        __bf16* dst = Bsb + tid * LDS_STRIDE;
#pragma unroll
        for (int h = 0; h < 2; ++h) {
            uint4 pk = *(const uint4*)(wRow + (kk >> 1) + h * 16);   // 32 nibbles
            const unsigned dw[4] = {pk.x, pk.y, pk.z, pk.w};
#pragma unroll
            for (int q = 0; q < 4; ++q) {
                uint4 o;
                unsigned w = dw[q];
#pragma unroll
                for (int j = 0; j < 4; ++j) {
                    float lo = (float)((w >> (8 * j)) & 15u) * s + mn;
                    float hi = (float)((w >> (8 * j + 4)) & 15u) * s + mn;
                    ((unsigned*)&o)[j] = pack2_bf16(lo, hi);
                }
                *(uint4*)(dst + h * 32 + q * 8) = o;
            }
        }
    }
}

__global__ __launch_bounds__(256)
void qlinear_wmma_bf16_kernel(const float* __restrict__ x,
                              const unsigned char* __restrict__ wpk,
                              const _Float16* __restrict__ wrange,
                              const _Float16* __restrict__ wmin,
                              float* __restrict__ out) {
    // double-buffered tiles: 2*(128*72 + 256*72)*2B = 108 KB (< 320 KB/WGP)
    __shared__ __bf16 As[2][BM * LDS_STRIDE];   // [m][k] x tile, bf16
    __shared__ __bf16 Bs[2][BN * LDS_STRIDE];   // [n][k] dequantized W tile, bf16

    const int tid   = threadIdx.x;
    const int lane  = tid & 31;
    const int wave  = tid >> 5;            // 8 waves: 2 (M) x 4 (N)
    const int wm    = wave >> 2;           // 0..1 -> 64-row strips
    const int wn    = wave & 3;            // 0..3 -> 64-col strips
    const int m0w   = wm * 64;
    const int n0w   = wn * 64;
    const int khalf = lane >> 4;
    const int l16   = lane & 15;

    const int blockN = blockIdx.x;         // 43 tiles of 256 out-features
    const int blockM = blockIdx.y;         // 64 tiles of 128 tokens
    const int mBase  = blockM * BM;
    const int nBase  = blockN * BN;

    // staging: A -> 2 threads/row (32 k each); B -> 1 thread/row (64 k)
    const int ldRow  = tid >> 1;
    const int ldKs   = (tid & 1) * 32;

    const float* xRow = x + (size_t)(mBase + ldRow) * IN_FEATURES;
    const int    oRow = nBase + tid;
    const unsigned char* wRow = wpk + (size_t)oRow * (IN_FEATURES / 2);

    v8f acc[4][4];
#pragma unroll
    for (int i = 0; i < 4; ++i)
#pragma unroll
        for (int j = 0; j < 4; ++j) acc[i][j] = (v8f)(0.0f);

    // prologue: stage chunk 0 into buffer 0
    stage_tiles(As[0], Bs[0], xRow, wRow, wrange, wmin, oRow, ldRow, ldKs, tid, 0);

#pragma unroll 1
    for (int s = 0; s < NSTAGES; ++s) {
        const int buf = s & 1;
        __syncthreads();   // staged data for buf ready; buf^1 free for reuse

        // stage next chunk into the other buffer; global loads + dequant VALU
        // co-execute with this stage's WMMA chain (XDL/VALU co-exec)
        if (s + 1 < NSTAGES) {
            stage_tiles(As[buf ^ 1], Bs[buf ^ 1], xRow, wRow, wrange, wmin,
                        oRow, ldRow, ldKs, tid, (s + 1) * BK);
            if (s + 2 < NSTAGES) {  // global_prefetch_b8 for stage s+2
                __builtin_prefetch(xRow + (s + 2) * BK + ldKs, 0, 1);
                __builtin_prefetch(wRow + (((s + 2) * BK) >> 1), 0, 1);
            }
        }

        // ---- compute: 2 k-steps x (4 M x 4 N) WMMAs per wave ----
#pragma unroll
        for (int kb = 0; kb < BK; kb += 32) {
            v16bf a[4], b[4];
#pragma unroll
            for (int tm = 0; tm < 4; ++tm) {
                const __bf16* row = &As[buf][(m0w + tm * 16 + l16) * LDS_STRIDE];
                a[tm] = load_frag_a(row, kb, khalf);
            }
#pragma unroll
            for (int tn = 0; tn < 4; ++tn) {
                const __bf16* row = &Bs[buf][(n0w + tn * 16 + l16) * LDS_STRIDE];
                b[tn] = load_frag_b(row, kb, khalf);
            }
#pragma unroll
            for (int tm = 0; tm < 4; ++tm)
#pragma unroll
                for (int tn = 0; tn < 4; ++tn)
                    acc[tm][tn] = __builtin_amdgcn_wmma_f32_16x16x32_bf16(
                        false, a[tm], false, b[tn], (short)0, acc[tm][tn],
                        false, false);
        }
    }

    // ---- writeback: C layout lane<16 -> M=r, lane>=16 -> M=r+8; N = lane&15 ----
#pragma unroll
    for (int tm = 0; tm < 4; ++tm) {
#pragma unroll
        for (int tn = 0; tn < 4; ++tn) {
            const int n_g = nBase + n0w + tn * 16 + l16;
#pragma unroll
            for (int r = 0; r < 8; ++r) {
                const int m_g = mBase + m0w + tm * 16 + r + khalf * 8;
                out[(size_t)m_g * OUT_FEATURES + n_g] = acc[tm][tn][r];
            }
        }
    }
}

extern "C" void kernel_launch(void* const* d_in, const int* in_sizes, int n_in,
                              void* d_out, int out_size, void* d_ws, size_t ws_size,
                              hipStream_t stream) {
    (void)in_sizes; (void)n_in; (void)out_size; (void)d_ws; (void)ws_size;
    const float*         x  = (const float*)d_in[0];
    const unsigned char* wp = (const unsigned char*)d_in[1];
    const _Float16*      wr = (const _Float16*)d_in[2];
    const _Float16*      wm = (const _Float16*)d_in[3];
    float*               o  = (float*)d_out;

    dim3 grid(OUT_FEATURES / BN, (4 * 2048) / BM);   // 43 x 64 tiles
    dim3 block(256);
    qlinear_wmma_bf16_kernel<<<grid, block, 0, stream>>>(x, wp, wr, wm, o);
}
